// LlamaGenAttention_5093831213409
// MI455X (gfx1250) — compile-verified, matrix operand order checked by reading, not run on hardware
//
#include <hip/hip_runtime.h>
#include <hip/hip_bf16.h>

typedef unsigned short u16;
typedef __attribute__((ext_vector_type(16))) __bf16 v16bf;
typedef __attribute__((ext_vector_type(8)))  __bf16 v8bf;
typedef __attribute__((ext_vector_type(8)))  float  v8f;
typedef int v4i_a __attribute__((vector_size(4 * sizeof(int))));

#define DIMC  2048
#define NHEAD 32
#define NKV   8
#define HD    64
#define BB    2
#define SSEQ  2048
#define QKVN  ((NHEAD + 2 * NKV) * HD)   // 3072
#define MTOT  (BB * SSEQ)                // 4096

#if __has_builtin(__builtin_amdgcn_global_load_async_to_lds_b128)
#define HAVE_ASYNC_LDS 1
#else
#define HAVE_ASYNC_LDS 0
#endif

// ---------- bf16 helpers ----------
__device__ __forceinline__ u16 f2bfb(float f) {
  return __builtin_bit_cast(u16, __float2bfloat16(f));
}
__device__ __forceinline__ float bf2f(u16 u) {
  return __bfloat162float(__builtin_bit_cast(__hip_bfloat16, u));
}

__device__ __forceinline__ v8f wmma_bf16(v16bf a, v16bf b, v8f c) {
  // (neg_a, A, neg_b, B, c_mod, C, reuse_a, reuse_b)
  return __builtin_amdgcn_wmma_f32_16x16x32_bf16(false, a, false, b, (short)0, c,
                                                 false, false);
}

// Load a v16bf A/B fragment whose 32-element K window starts at kwin.
// ISA 7.12.2 16-bit layout: lanes 0-15 hold K {0..7,16..23}, lanes 16-31 hold
// K {8..15,24..31} -> per-lane two contiguous 8-element runs.
__device__ __forceinline__ v16bf load_frag(const u16* kwin, int lane) {
  const int h = (lane >> 4) << 3;
  v8bf lo = *reinterpret_cast<const v8bf*>(kwin + h);
  v8bf hi = *reinterpret_cast<const v8bf*>(kwin + 16 + h);
  v16bf r;
#pragma unroll
  for (int i = 0; i < 8; ++i) { r[i] = lo[i]; r[i + 8] = hi[i]; }
  return r;
}

__device__ __forceinline__ void wait_async_lds() {
#if __has_builtin(__builtin_amdgcn_s_wait_asynccnt)
  __builtin_amdgcn_s_wait_asynccnt(0);
#else
  asm volatile("s_wait_asynccnt 0x0" ::: "memory");
#endif
}

// ---------- kernel 0: fp32 -> bf16 ----------
__global__ void f32_to_bf16_kernel(const float* __restrict__ in,
                                   u16* __restrict__ out, size_t n) {
  size_t i = (size_t)blockIdx.x * blockDim.x + threadIdx.x;
  if (i < n) out[i] = f2bfb(in[i]);
}

// ---------- GEMM: C = A(MxK) * B(KxN), bf16 in, f32 acc ----------
// Block: 128 threads (4 waves). Block tile 128x64, wave tile 32x64.
template <bool OUT_BF16>
__global__ __launch_bounds__(128) void gemm_bf16_kernel(
    const u16* __restrict__ A, const u16* __restrict__ B, void* __restrict__ C,
    int M, int N, int K) {
  __shared__ u16 As[128 * 32];      // row-major 128 rows x 32 k (8 KB)
  __shared__ u16 BsT[64 * 40];      // transposed: 64 n-rows x 32 k, padded (5 KB)
  const int tid  = threadIdx.x;
  const int lane = tid & 31;
  const int wave = tid >> 5;
  const int m0 = blockIdx.y * 128;
  const int n0 = blockIdx.x * 64;

  v8f acc[2][4] = {};

  for (int k0 = 0; k0 < K; k0 += 32) {
    // ---- Stage A tile: thread t -> full 32-wide K row for M row t ----
    {
      const int r = tid;  // 0..127
      const u16* src = A + (size_t)(m0 + r) * K + k0;
#if HAVE_ASYNC_LDS
#pragma unroll
      for (int c = 0; c < 32; c += 8)
        __builtin_amdgcn_global_load_async_to_lds_b128(
            (v4i_a*)(src + c), (v4i_a*)&As[r * 32 + c], 0, 0);
#else
#pragma unroll
      for (int c = 0; c < 32; c += 8)
        *reinterpret_cast<v8bf*>(&As[r * 32 + c]) =
            *reinterpret_cast<const v8bf*>(src + c);
#endif
      if (k0 + 32 < K) __builtin_prefetch(src + 32, 0, 1);
    }
    // ---- Stage B tile transposed: thread t -> k row t/4, 16 n at (t&3)*16 ----
    {
      const int kk = tid >> 2;
      const int nb = (tid & 3) * 16;
      const u16* src = B + (size_t)(k0 + kk) * N + n0 + nb;
#pragma unroll
      for (int i = 0; i < 16; ++i) BsT[(nb + i) * 40 + kk] = src[i];
      if (k0 + 32 < K) __builtin_prefetch(src + (size_t)32 * N, 0, 1);
    }
#if HAVE_ASYNC_LDS
    wait_async_lds();
#endif
    __syncthreads();

    const int ar0 = wave * 32 + (lane & 15);
    v16bf af0 = load_frag(&As[ar0 * 32], lane);
    v16bf af1 = load_frag(&As[(ar0 + 16) * 32], lane);
#pragma unroll
    for (int t = 0; t < 4; ++t) {
      const int n = t * 16 + (lane & 15);
      v16bf bfrag = load_frag(&BsT[n * 40], lane);
      acc[0][t] = wmma_bf16(af0, bfrag, acc[0][t]);
      acc[1][t] = wmma_bf16(af1, bfrag, acc[1][t]);
    }
    __syncthreads();
  }

  // Epilogue: C-fragment layout -> lane holds col (lane&15), rows r + 8*(lane>=16)
#pragma unroll
  for (int s = 0; s < 2; ++s) {
#pragma unroll
    for (int t = 0; t < 4; ++t) {
#pragma unroll
      for (int r = 0; r < 8; ++r) {
        const int row = m0 + wave * 32 + s * 16 + ((lane >> 4) << 3) + r;
        const int col = n0 + t * 16 + (lane & 15);
        if (OUT_BF16)
          ((u16*)C)[(size_t)row * N + col] = f2bfb(acc[s][t][r]);
        else
          ((float*)C)[(size_t)row * N + col] = acc[s][t][r];
      }
    }
  }
}

// ---------- RoPE + head split (+ fold 1/sqrt(d) into Q) ----------
// qkv: (B,S,48,64) bf16. Outputs: q (B,32,S,64), k (B,8,S,64),
// vt TRANSPOSED (B,8,64,S) so the P*V B-fragment loads are contiguous runs
// along the key axis.
__global__ void rope_split_kernel(const u16* __restrict__ qkv,
                                  const float* __restrict__ fc,
                                  u16* __restrict__ q, u16* __restrict__ k,
                                  u16* __restrict__ vt) {
  const size_t tid = (size_t)blockIdx.x * blockDim.x + threadIdx.x;
  const int p    = (int)(tid & 31);          // pair index within head_dim/2
  const int head = (int)((tid >> 5) % 48);
  const int s    = (int)(((tid >> 5) / 48) % SSEQ);
  const int b    = (int)((tid >> 5) / 48 / SSEQ);

  const u16* src = qkv + ((size_t)(b * SSEQ + s) * QKVN) + head * HD + 2 * p;
  float x0 = bf2f(src[0]);
  float x1 = bf2f(src[1]);

  if (head < NHEAD + NKV) {  // rotary for q and k heads
    const float f0 = fc[((size_t)s * 32 + p) * 2 + 0];
    const float f1 = fc[((size_t)s * 32 + p) * 2 + 1];
    const float re = x0 * f0 - x1 * f1;
    const float im = x1 * f0 + x0 * f1;
    x0 = re; x1 = im;
  }

  if (head < NHEAD) {
    const float sc = 0.125f;  // 1/sqrt(64), folded into Q
    u16* dst = q + (((size_t)(b * NHEAD + head) * SSEQ + s) * HD) + 2 * p;
    dst[0] = f2bfb(x0 * sc);
    dst[1] = f2bfb(x1 * sc);
  } else if (head < NHEAD + NKV) {
    const int hh = head - NHEAD;
    u16* dst = k + (((size_t)(b * NKV + hh) * SSEQ + s) * HD) + 2 * p;
    dst[0] = f2bfb(x0);
    dst[1] = f2bfb(x1);
  } else {
    const int hh = head - NHEAD - NKV;
    // vt[b][hh][d][s], d = 2p and 2p+1
    u16* base = vt + ((size_t)(b * NKV + hh) * HD) * SSEQ + s;
    base[(size_t)(2 * p) * SSEQ]     = f2bfb(x0);
    base[(size_t)(2 * p + 1) * SSEQ] = f2bfb(x1);
  }
}

// ---------- Flash attention: 1 wave per (b, head, 16-row q tile) ----------
__global__ __launch_bounds__(32) void flash_kernel(const u16* __restrict__ q,
                                                   const u16* __restrict__ k,
                                                   const u16* __restrict__ vt,
                                                   u16* __restrict__ o) {
  const int lane = threadIdx.x;
  const int i0 = blockIdx.x * 16;
  const int h  = blockIdx.y;
  const int b  = blockIdx.z;
  const int kh = h >> 2;  // GROUP = 4

  const u16* qbase  = q  + ((size_t)(b * NHEAD + h) * SSEQ + i0) * HD;
  const u16* kbase  = k  + ((size_t)(b * NKV + kh) * SSEQ) * HD;
  const u16* vtbase = vt + ((size_t)(b * NKV + kh) * HD) * SSEQ;

  // Q fragments (M=16, K=64 -> two 16x32 A fragments); scale pre-folded.
  const int qrow = lane & 15;
  const v16bf aq0 = load_frag(qbase + qrow * HD + 0, lane);
  const v16bf aq1 = load_frag(qbase + qrow * HD + 32, lane);

  v8f acc[4] = {};
  float m[8], l[8];
#pragma unroll
  for (int r = 0; r < 8; ++r) { m[r] = -1e30f; l[r] = 0.0f; }

  __shared__ u16 psh[16 * 32];  // P tile staging (C-layout -> A-layout)

  for (int j0 = 0; j0 <= i0 + 15; j0 += 32) {
    // ---- scores: S(16x32) = Q(16x64) * K(j0..j0+31, :)^T ----
    v8f s0 = {}, s1 = {};
    {
      const int kn = lane & 15;
      const u16* krow0 = kbase + (size_t)(j0 + kn) * HD;
      const u16* krow1 = kbase + (size_t)(j0 + 16 + kn) * HD;
      v16bf b00 = load_frag(krow0 + 0, lane);
      v16bf b01 = load_frag(krow0 + 32, lane);
      v16bf b10 = load_frag(krow1 + 0, lane);
      v16bf b11 = load_frag(krow1 + 32, lane);
      s0 = wmma_bf16(aq0, b00, s0);
      s0 = wmma_bf16(aq1, b01, s0);
      s1 = wmma_bf16(aq0, b10, s1);
      s1 = wmma_bf16(aq1, b11, s1);
    }

    // ---- causal mask + online softmax ----
    const int col0 = j0 + (lane & 15);
    const int col1 = col0 + 16;
    float tm[8];
#pragma unroll
    for (int r = 0; r < 8; ++r) {
      const int grow = i0 + ((lane >> 4) << 3) + r;
      float x0 = (col0 <= grow) ? s0[r] : -1e30f;
      float x1 = (col1 <= grow) ? s1[r] : -1e30f;
      s0[r] = x0; s1[r] = x1;
      tm[r] = fmaxf(x0, x1);
    }
#pragma unroll
    for (int off = 1; off < 16; off <<= 1) {
#pragma unroll
      for (int r = 0; r < 8; ++r)
        tm[r] = fmaxf(tm[r], __shfl_xor(tm[r], off, 32));
    }

    float alpha[8], rs[8];
#pragma unroll
    for (int r = 0; r < 8; ++r) {
      const float mn = fmaxf(m[r], tm[r]);
      alpha[r] = __expf(m[r] - mn);
      m[r] = mn;
      const float p0 = __expf(s0[r] - mn);
      const float p1 = __expf(s1[r] - mn);
      s0[r] = p0; s1[r] = p1;
      rs[r] = p0 + p1;
    }
#pragma unroll
    for (int off = 1; off < 16; off <<= 1) {
#pragma unroll
      for (int r = 0; r < 8; ++r) rs[r] += __shfl_xor(rs[r], off, 32);
    }
#pragma unroll
    for (int r = 0; r < 8; ++r) l[r] = l[r] * alpha[r] + rs[r];
#pragma unroll
    for (int t = 0; t < 4; ++t)
#pragma unroll
      for (int r = 0; r < 8; ++r) acc[t][r] *= alpha[r];

    // ---- P (C-layout f32) -> LDS -> A-layout bf16 fragment ----
    __syncthreads();
#pragma unroll
    for (int r = 0; r < 8; ++r) {
      const int row = ((lane >> 4) << 3) + r;
      psh[row * 32 + (lane & 15)]      = f2bfb(s0[r]);
      psh[row * 32 + 16 + (lane & 15)] = f2bfb(s1[r]);
    }
    __syncthreads();
    const v16bf ap = load_frag(&psh[(lane & 15) * 32], lane);

    // ---- O += P(16x32) * V(32x64): V^T rows are contiguous along keys ----
#pragma unroll
    for (int t = 0; t < 4; ++t) {
      const int dcol = t * 16 + (lane & 15);
      v16bf bv = load_frag(vtbase + (size_t)dcol * SSEQ + j0, lane);
      acc[t] = wmma_bf16(ap, bv, acc[t]);
    }
    __syncthreads();
  }

  // ---- finalize + store to (B,S,DIM) bf16 ----
#pragma unroll
  for (int t = 0; t < 4; ++t) {
#pragma unroll
    for (int r = 0; r < 8; ++r) {
      const int row = ((lane >> 4) << 3) + r;
      const int srow = i0 + row;
      const float ov = acc[t][r] / l[r];
      o[((size_t)b * SSEQ + srow) * DIMC + h * HD + t * 16 + (lane & 15)] =
          f2bfb(ov);
    }
  }
}

extern "C" void kernel_launch(void* const* d_in, const int* in_sizes, int n_in,
                              void* d_out, int out_size, void* d_ws,
                              size_t ws_size, hipStream_t stream) {
  (void)in_sizes; (void)n_in; (void)out_size; (void)ws_size;
  const float* x    = (const float*)d_in[0];
  const float* fc   = (const float*)d_in[1];
  const float* wqkv = (const float*)d_in[2];
  const float* wo   = (const float*)d_in[3];
  float* out = (float*)d_out;

  // Workspace layout (bf16 elements), ~105 MB total.
  u16* ws = (u16*)d_ws;
  size_t off = 0;
  u16* xw    = ws + off; off += (size_t)MTOT * DIMC;        // x bf16
  u16* wqkvw = ws + off; off += (size_t)DIMC * QKVN;        // Wqkv bf16
  u16* wow   = ws + off; off += (size_t)DIMC * DIMC;        // Wo bf16
  u16* qkv   = ws + off; off += (size_t)MTOT * QKVN;        // qkv bf16
  u16* qh    = ws + off; off += (size_t)BB * NHEAD * SSEQ * HD;
  u16* khb   = ws + off; off += (size_t)BB * NKV * SSEQ * HD;
  u16* vtb   = ws + off; off += (size_t)BB * NKV * SSEQ * HD;  // V transposed
  u16* attn  = ws + off; off += (size_t)MTOT * DIMC;        // attention out bf16

  // 1) fp32 -> bf16 conversions
  {
    size_t n;
    n = (size_t)MTOT * DIMC;
    f32_to_bf16_kernel<<<dim3((unsigned)((n + 255) / 256)), dim3(256), 0, stream>>>(x, xw, n);
    n = (size_t)DIMC * QKVN;
    f32_to_bf16_kernel<<<dim3((unsigned)((n + 255) / 256)), dim3(256), 0, stream>>>(wqkv, wqkvw, n);
    n = (size_t)DIMC * DIMC;
    f32_to_bf16_kernel<<<dim3((unsigned)((n + 255) / 256)), dim3(256), 0, stream>>>(wo, wow, n);
  }

  // 2) QKV projection: (4096x2048) * (2048x3072) -> bf16
  gemm_bf16_kernel<true><<<dim3(QKVN / 64, MTOT / 128), dim3(128), 0, stream>>>(
      xw, wqkvw, (void*)qkv, MTOT, QKVN, DIMC);

  // 3) RoPE + split into head-major Q/K and transposed V
  {
    const size_t nthreads = (size_t)BB * SSEQ * 48 * 32;  // 6,291,456
    rope_split_kernel<<<dim3((unsigned)(nthreads / 256)), dim3(256), 0, stream>>>(
        qkv, fc, qh, khb, vtb);
  }

  // 4) Flash attention (causal, GQA): 1 wave per 16-row q tile per head
  flash_kernel<<<dim3(SSEQ / 16, NHEAD, BB), dim3(32), 0, stream>>>(qh, khb, vtb, attn);

  // 5) Output projection: (4096x2048) * (2048x2048) -> f32 d_out
  gemm_bf16_kernel<false><<<dim3(DIMC / 64, MTOT / 128), dim3(128), 0, stream>>>(
      attn, wow, (void*)out, MTOT, DIMC, DIMC);
}
